// ProteinGrid_66108136620512
// MI455X (gfx1250) — matrix operand order repaired and checked by reading, not compiled, not run on hardware
//
#include <hip/hip_runtime.h>
#include <hip/hip_bf16.h>

typedef __attribute__((ext_vector_type(16))) _Float16 v16h;
typedef __attribute__((ext_vector_type(8)))  _Float16 v8h;
typedef __attribute__((ext_vector_type(4)))  _Float16 v4h;
typedef __attribute__((ext_vector_type(8)))  float    v8f;

// ---------------------------------------------------------------------------
// WMMA helpers (CDNA5 V_WMMA_F32_16X16X32_F16)
// C/D layout: lane L, reg r -> (M = r + 8*(L>>4), N = L&15)
// A layout  : lane L, half h -> M = L&15, K = (h&7) + 8*((h>>3)*2 + (L>>4))
//             i.e. two contiguous 8-half chunks at K = 8*(L>>4) and 16+8*(L>>4)
// B layout  : lane L, half h -> N = L&15, K = (h&7) + 8*((h>>3)*2 + (L>>4))  (prepacked)
// ---------------------------------------------------------------------------
__device__ __forceinline__ v8f wmma32(v16h a, v16h b, v8f c) {
  return __builtin_amdgcn_wmma_f32_16x16x32_f16(false, a, false, b, (short)0, c,
                                                false, false);
}

__device__ __forceinline__ v16h load_a_frag(const _Float16* p, int lane) {
  const int off = (lane >> 4) * 8;
  v8h lo = *(const v8h*)(p + off);
  v8h hh = *(const v8h*)(p + off + 16);
  v16h a;
#pragma unroll
  for (int i = 0; i < 8; ++i) { a[i] = lo[i]; a[i + 8] = hh[i]; }
  return a;
}

__device__ __forceinline__ v16h load_b_frag(const _Float16* pw, size_t frag,
                                            int lane) {
  return *(const v16h*)(pw + frag * 512 + (size_t)lane * 16);
}

// fast SiLU: hardware v_rcp_f32 instead of IEEE division (inputs are f16
// anyway)
__device__ __forceinline__ float silu(float x) {
  return x * __builtin_amdgcn_rcpf(1.f + __expf(-x));
}

// ---------------------------------------------------------------------------
// Weight packing: f32 -> f16 B fragments (512 halfs / fragment, 1 frag/block)
// ---------------------------------------------------------------------------
__global__ __launch_bounds__(32) void pack_lin_b(const float* __restrict__ W,
                                                 _Float16* __restrict__ dst,
                                                 int Krows, int Ncols, int nbC) {
  const int frag = blockIdx.x;
  const int kc = frag / nbC, nb = frag % nbC;
  const int lane = threadIdx.x & 31;
  _Float16* out = dst + (size_t)frag * 512 + lane * 16;
  const int n = nb * 16 + (lane & 15);
  const int kbase = kc * 32 + ((lane >> 4) << 3);
  for (int h = 0; h < 16; ++h) {
    const int k = kbase + (h & 7) + ((h >> 3) << 4);
    out[h] = (_Float16)((k < Krows) ? W[(size_t)k * Ncols + n] : 0.f);
  }
}

// conv weights W[O][I][T] (T = 27 or 125), frag index = (t*kcC + kc)*nbC + nb
__global__ __launch_bounds__(32) void pack_conv_b(const float* __restrict__ W,
                                                  _Float16* __restrict__ dst,
                                                  int I, int O, int T, int kcC,
                                                  int nbC) {
  const int frag = blockIdx.x;
  const int nb = frag % nbC;
  const int tmp = frag / nbC;
  const int kc = tmp % kcC;
  const int t = tmp / kcC;
  const int lane = threadIdx.x & 31;
  _Float16* out = dst + (size_t)frag * 512 + lane * 16;
  const int co = nb * 16 + (lane & 15);
  const int kbase = kc * 32 + ((lane >> 4) << 3);
  for (int h = 0; h < 16; ++h) {
    const int ci = kbase + (h & 7) + ((h >> 3) << 4);
    out[h] = (_Float16)W[((size_t)co * I + ci) * T + t];
  }
}

// ---------------------------------------------------------------------------
// Kernel 1: MPNN edge+message MLPs + segment-mean. One block per voxel j,
// edges j*256..j*256+255 are contiguous (edge_j = repeat(arange, 256)).
// 4 waves, each wave runs 4 tiles of 16 edges through 4 chained WMMA GEMMs.
// ---------------------------------------------------------------------------
__global__ __launch_bounds__(128) void mpnn_edge_kernel(
    const float* __restrict__ node_emb, const float* __restrict__ node_pos,
    const float* __restrict__ grid_pos, const int* __restrict__ edge_i,
    const _Float16* __restrict__ pwEW1, const _Float16* __restrict__ pwEW2,
    const _Float16* __restrict__ pwMW1, const _Float16* __restrict__ pwMW2,
    const float* __restrict__ eb1, const float* __restrict__ eb2,
    const float* __restrict__ mb1, const float* __restrict__ mb2,
    _Float16* __restrict__ mean_out) {
  __shared__ __align__(16) _Float16 stA[4][16 * 128];  // edge-attr / concat in
  __shared__ __align__(16) _Float16 stH[4][16 * 64];   // hidden staging
  __shared__ float acc[64];
  const int wave = threadIdx.x >> 5, lane = threadIdx.x & 31;
  const int j = blockIdx.x;
  const int m = lane & 15, hi = lane >> 4;
  if (threadIdx.x < 64) acc[threadIdx.x] = 0.f;
  __syncthreads();
  const float gp0 = grid_pos[j * 3 + 0];
  const float gp1 = grid_pos[j * 3 + 1];
  const float gp2 = grid_pos[j * 3 + 2];
  _Float16* A2 = stA[wave];
  _Float16* H = stH[wave];
  for (int tt = wave; tt < 16; tt += 4) {
    const int e0 = j * 256 + tt * 16;
    // stage edge_attr [16 x 6 pad 32] into A2 cols 0..31
    if (lane < 16) {
      const int ii = edge_i[e0 + lane];
      _Float16* rp = A2 + lane * 128;
      rp[0] = (_Float16)node_pos[ii * 3 + 0];
      rp[1] = (_Float16)node_pos[ii * 3 + 1];
      rp[2] = (_Float16)node_pos[ii * 3 + 2];
      rp[3] = (_Float16)gp0; rp[4] = (_Float16)gp1; rp[5] = (_Float16)gp2;
      for (int c = 6; c < 32; ++c) rp[c] = (_Float16)0.f;
    }
    __syncthreads();
    // e1: silu([16,32] @ ew1 + eb1) -> H
    for (int nb = 0; nb < 4; ++nb) {
      v16h a = load_a_frag(A2 + m * 128, lane);
      v16h b = load_b_frag(pwEW1, nb, lane);
      v8f d = {};
      d = wmma32(a, b, d);
      const float bias = eb1[nb * 16 + m];
#pragma unroll
      for (int r = 0; r < 8; ++r)
        H[(r + 8 * hi) * 64 + nb * 16 + m] = (_Float16)silu(d[r] + bias);
    }
    __syncthreads();
    // e2: pos_emb = H @ ew2 + eb2 -> A2 cols 64..127 ; stage emb cols 0..63
    for (int nb = 0; nb < 4; ++nb) {
      v8f d = {};
      for (int kc = 0; kc < 2; ++kc) {
        v16h a = load_a_frag(H + m * 64 + kc * 32, lane);
        v16h b = load_b_frag(pwEW2, kc * 4 + nb, lane);
        d = wmma32(a, b, d);
      }
      const float bias = eb2[nb * 16 + m];
#pragma unroll
      for (int r = 0; r < 8; ++r)
        A2[(r + 8 * hi) * 128 + 64 + nb * 16 + m] = (_Float16)(d[r] + bias);
    }
    {
      // stage node_embedding rows into A2 cols 0..63: all 32 lanes, half a
      // row (32 channels) each, float4 loads + packed 4xf16 stores
      const int rr = lane >> 1;
      const int c0 = (lane & 1) * 32;
      const int ii = edge_i[e0 + rr];
      const float* src = node_emb + (size_t)ii * 64 + c0;
      _Float16* rp = A2 + rr * 128 + c0;
#pragma unroll
      for (int c = 0; c < 32; c += 4) {
        const float4 f = *(const float4*)(src + c);
        v4h hv;
        hv[0] = (_Float16)f.x; hv[1] = (_Float16)f.y;
        hv[2] = (_Float16)f.z; hv[3] = (_Float16)f.w;
        *(v4h*)(rp + c) = hv;
      }
    }
    __syncthreads();
    // m1: silu([16,128] @ mw1 + mb1) -> H
    for (int nb = 0; nb < 4; ++nb) {
      v8f d = {};
      for (int kc = 0; kc < 4; ++kc) {
        v16h a = load_a_frag(A2 + m * 128 + kc * 32, lane);
        v16h b = load_b_frag(pwMW1, kc * 4 + nb, lane);
        d = wmma32(a, b, d);
      }
      const float bias = mb1[nb * 16 + m];
#pragma unroll
      for (int r = 0; r < 8; ++r)
        H[(r + 8 * hi) * 64 + nb * 16 + m] = (_Float16)silu(d[r] + bias);
    }
    __syncthreads();
    // m2: msg = H @ mw2 (+mb2 folded into mean); accumulate column sums
    for (int nb = 0; nb < 4; ++nb) {
      v8f d = {};
      for (int kc = 0; kc < 2; ++kc) {
        v16h a = load_a_frag(H + m * 64 + kc * 32, lane);
        v16h b = load_b_frag(pwMW2, kc * 4 + nb, lane);
        d = wmma32(a, b, d);
      }
      float s = 0.f;
#pragma unroll
      for (int r = 0; r < 8; ++r) s += d[r];
      atomicAdd(&acc[nb * 16 + m], s);
    }
    __syncthreads();
  }
  if (threadIdx.x < 64) {
    mean_out[(size_t)j * 64 + threadIdx.x] =
        (_Float16)(acc[threadIdx.x] * (1.f / 256.f) + mb2[threadIdx.x]);
  }
}

// ---------------------------------------------------------------------------
// Kernel 2: update MLP over [4096,64], output x0 = (mlp)*BN as f16 [vox][64]
// ---------------------------------------------------------------------------
__global__ __launch_bounds__(128) void update_kernel(
    const _Float16* __restrict__ mean16, const _Float16* __restrict__ pwUW1,
    const _Float16* __restrict__ pwUW2, const float* __restrict__ ub1,
    const float* __restrict__ ub2, _Float16* __restrict__ x0, float scale) {
  __shared__ __align__(16) _Float16 stH[4][16 * 64];
  const int wave = threadIdx.x >> 5, lane = threadIdx.x & 31;
  const int m = lane & 15, hi = lane >> 4;
  const int v0 = (blockIdx.x * 4 + wave) * 16;
  _Float16* H = stH[wave];
  const _Float16* Arow = mean16 + (size_t)(v0 + m) * 64;
  for (int nb = 0; nb < 4; ++nb) {
    v8f d = {};
    for (int kc = 0; kc < 2; ++kc) {
      v16h a = load_a_frag(Arow + kc * 32, lane);
      v16h b = load_b_frag(pwUW1, kc * 4 + nb, lane);
      d = wmma32(a, b, d);
    }
    const float bias = ub1[nb * 16 + m];
#pragma unroll
    for (int r = 0; r < 8; ++r)
      H[(r + 8 * hi) * 64 + nb * 16 + m] = (_Float16)silu(d[r] + bias);
  }
  __syncthreads();
  for (int nb = 0; nb < 4; ++nb) {
    v8f d = {};
    for (int kc = 0; kc < 2; ++kc) {
      v16h a = load_a_frag(H + m * 64 + kc * 32, lane);
      v16h b = load_b_frag(pwUW2, kc * 4 + nb, lane);
      d = wmma32(a, b, d);
    }
    const float bias = ub2[nb * 16 + m];
#pragma unroll
    for (int r = 0; r < 8; ++r)
      x0[(size_t)(v0 + r + 8 * hi) * 64 + nb * 16 + m] =
          (_Float16)((d[r] + bias) * scale);
  }
}

// ---------------------------------------------------------------------------
// Implicit-GEMM conv3d over 16^3 grid. One wave = one (z,y) row (16 x-voxels,
// the M tile) x one 16-wide Cout block. Taps accumulate into one f32 WMMA acc.
// ---------------------------------------------------------------------------
__device__ __forceinline__ void conv_acc(v8f& d, const _Float16* __restrict__ in,
                                         const _Float16* __restrict__ pw,
                                         int Cin, int k, int z, int y, int nb,
                                         int nbC, int lane) {
  const int kcC = Cin >> 5;
  const int m = lane & 15;
  const int koff = (lane >> 4) * 8;
  int t = 0;
  for (int dz = -k; dz <= k; ++dz) {
    const int zz = z + dz;
    for (int dy = -k; dy <= k; ++dy) {
      const int yy = y + dy;
      for (int dx = -k; dx <= k; ++dx, ++t) {
        if ((unsigned)zz >= 16u || (unsigned)yy >= 16u) continue;  // zero pad
        const int xx = m + dx;
        const bool inb = (unsigned)xx < 16u;
        const _Float16* rowp =
            in + (size_t)((zz * 16 + yy) * 16 + (inb ? xx : 0)) * Cin;
        const _Float16* fp =
            pw + ((size_t)t * kcC * nbC + nb) * 512 + (size_t)lane * 16;
        for (int kc = 0; kc < kcC; ++kc) {
          v16h a;
          if (inb) {
            const _Float16* p = rowp + kc * 32 + koff;
            v8h lo = *(const v8h*)p;
            v8h hh = *(const v8h*)(p + 16);
#pragma unroll
            for (int i2 = 0; i2 < 8; ++i2) { a[i2] = lo[i2]; a[i2 + 8] = hh[i2]; }
          } else {
#pragma unroll
            for (int i2 = 0; i2 < 16; ++i2) a[i2] = (_Float16)0.f;
          }
          v16h b = *(const v16h*)fp;
          fp += (size_t)nbC * 512;
          d = wmma32(a, b, d);
        }
      }
    }
  }
}

__global__ __launch_bounds__(128) void conv_kernel(
    const _Float16* __restrict__ inA, const _Float16* __restrict__ pwA,
    int CinA, int kA, const _Float16* __restrict__ inB,
    const _Float16* __restrict__ pwB, int CinB, int kB, int Cout,
    _Float16* __restrict__ out, float scale) {
  const int wave = threadIdx.x >> 5, lane = threadIdx.x & 31;
  const int nbC = Cout >> 4;
  const int task = blockIdx.x * 4 + wave;
  const int nb = task % nbC;
  const int row = task / nbC;  // 0..255 = (z,y)
  const int z = row >> 4, y = row & 15;
  const int m = lane & 15, hi = lane >> 4;
  v8f d = {};
  conv_acc(d, inA, pwA, CinA, kA, z, y, nb, nbC, lane);
  if (inB) conv_acc(d, inB, pwB, CinB, kB, z, y, nb, nbC, lane);  // shortcut
#pragma unroll
  for (int r = 0; r < 8; ++r) {
    float v = d[r] * scale;
    v = v > 0.f ? v : 0.f;  // relu (both block positions are relu'd)
    out[(size_t)((z * 16 + y) * 16 + r + 8 * hi) * Cout + nb * 16 + m] =
        (_Float16)v;
  }
}

// ---------------------------------------------------------------------------
// Global max-pool over 4096 voxels per channel + fc 256->20
// ---------------------------------------------------------------------------
__global__ __launch_bounds__(256) void pool_fc_kernel(
    const _Float16* __restrict__ y3, const float* __restrict__ fcw,
    const float* __restrict__ fcb, float* __restrict__ out) {
  __shared__ float pooled[256];
  const int c = threadIdx.x;
  float mx = -3.4e38f;
  for (int v = 0; v < 4096; ++v)
    mx = fmaxf(mx, (float)y3[(size_t)v * 256 + c]);
  pooled[c] = mx;
  __syncthreads();
  if (c < 20) {
    float s = fcb[c];
    for (int i = 0; i < 256; ++i) s += pooled[i] * fcw[i * 20 + c];
    out[c] = s;
  }
}

// ---------------------------------------------------------------------------
extern "C" void kernel_launch(void* const* d_in, const int* in_sizes, int n_in,
                              void* d_out, int out_size, void* d_ws,
                              size_t ws_size, hipStream_t stream) {
  (void)in_sizes; (void)n_in; (void)out_size; (void)ws_size;
  const float* node_emb = (const float*)d_in[0];
  const float* node_pos = (const float*)d_in[1];
  const float* grid_pos = (const float*)d_in[2];
  const int* edge_i = (const int*)d_in[3];  // row 0 of edge_index
  const float* ew1 = (const float*)d_in[4];  const float* eb1 = (const float*)d_in[5];
  const float* ew2 = (const float*)d_in[6];  const float* eb2 = (const float*)d_in[7];
  const float* mw1 = (const float*)d_in[8];  const float* mb1 = (const float*)d_in[9];
  const float* mw2 = (const float*)d_in[10]; const float* mb2 = (const float*)d_in[11];
  const float* uw1 = (const float*)d_in[12]; const float* ub1 = (const float*)d_in[13];
  const float* uw2 = (const float*)d_in[14]; const float* ub2 = (const float*)d_in[15];
  const float* c1a = (const float*)d_in[16]; const float* c1b = (const float*)d_in[17];
  const float* c1s = (const float*)d_in[18];
  const float* c2a = (const float*)d_in[19]; const float* c2b = (const float*)d_in[20];
  const float* c2s = (const float*)d_in[21];
  const float* c3a = (const float*)d_in[22]; const float* c3b = (const float*)d_in[23];
  const float* c3s = (const float*)d_in[24];
  const float* fcw = (const float*)d_in[25]; const float* fcb = (const float*)d_in[26];

  _Float16* wsh = (_Float16*)d_ws;
  size_t cur = 0;
  auto alloc = [&](size_t halfs) { _Float16* p = wsh + cur; cur += halfs; return p; };
  const size_t FR = 512;
  _Float16* pwEW1 = alloc(4 * FR);    _Float16* pwEW2 = alloc(8 * FR);
  _Float16* pwMW1 = alloc(16 * FR);   _Float16* pwMW2 = alloc(8 * FR);
  _Float16* pwUW1 = alloc(8 * FR);    _Float16* pwUW2 = alloc(8 * FR);
  _Float16* pwC1A = alloc(216 * FR);  _Float16* pwC1B = alloc(216 * FR);
  _Float16* pwC1S = alloc(1000 * FR);
  _Float16* pwC2A = alloc(432 * FR);  _Float16* pwC2B = alloc(864 * FR);
  _Float16* pwC2S = alloc(2000 * FR);
  _Float16* pwC3A = alloc(1728 * FR); _Float16* pwC3B = alloc(3456 * FR);
  _Float16* pwC3S = alloc(8000 * FR);
  _Float16* mean16 = alloc(4096 * 64);
  _Float16* x0 = alloc(4096 * 64);
  _Float16* t1 = alloc(4096 * 64);
  _Float16* y1 = alloc(4096 * 64);
  _Float16* t2 = alloc(4096 * 128);
  _Float16* y2 = alloc(4096 * 128);
  _Float16* t3 = alloc(4096 * 256);
  _Float16* y3 = alloc(4096 * 256);

  const float BN = 0.99999500003749977f;  // 1/sqrt(1+1e-5)

  // --- pack all weights into f16 WMMA B-fragments ---
  pack_lin_b<<<4, 32, 0, stream>>>(ew1, pwEW1, 6, 64, 4);
  pack_lin_b<<<8, 32, 0, stream>>>(ew2, pwEW2, 64, 64, 4);
  pack_lin_b<<<16, 32, 0, stream>>>(mw1, pwMW1, 128, 64, 4);
  pack_lin_b<<<8, 32, 0, stream>>>(mw2, pwMW2, 64, 64, 4);
  pack_lin_b<<<8, 32, 0, stream>>>(uw1, pwUW1, 64, 64, 4);
  pack_lin_b<<<8, 32, 0, stream>>>(uw2, pwUW2, 64, 64, 4);
  pack_conv_b<<<216, 32, 0, stream>>>(c1a, pwC1A, 64, 64, 27, 2, 4);
  pack_conv_b<<<216, 32, 0, stream>>>(c1b, pwC1B, 64, 64, 27, 2, 4);
  pack_conv_b<<<1000, 32, 0, stream>>>(c1s, pwC1S, 64, 64, 125, 2, 4);
  pack_conv_b<<<432, 32, 0, stream>>>(c2a, pwC2A, 64, 128, 27, 2, 8);
  pack_conv_b<<<864, 32, 0, stream>>>(c2b, pwC2B, 128, 128, 27, 4, 8);
  pack_conv_b<<<2000, 32, 0, stream>>>(c2s, pwC2S, 64, 128, 125, 2, 8);
  pack_conv_b<<<1728, 32, 0, stream>>>(c3a, pwC3A, 128, 256, 27, 4, 16);
  pack_conv_b<<<3456, 32, 0, stream>>>(c3b, pwC3B, 256, 256, 27, 8, 16);
  pack_conv_b<<<8000, 32, 0, stream>>>(c3s, pwC3S, 128, 256, 125, 4, 16);

  // --- MPNN: edge/message MLPs + segment-mean, then update MLP ---
  mpnn_edge_kernel<<<4096, 128, 0, stream>>>(node_emb, node_pos, grid_pos,
                                             edge_i, pwEW1, pwEW2, pwMW1, pwMW2,
                                             eb1, eb2, mb1, mb2, mean16);
  update_kernel<<<64, 128, 0, stream>>>(mean16, pwUW1, pwUW2, ub1, ub2, x0, BN);

  // --- ResNet3D blocks (implicit-GEMM WMMA convs, BN+relu fused) ---
  conv_kernel<<<256, 128, 0, stream>>>(x0, pwC1A, 64, 1, nullptr, nullptr, 0, 0,
                                       64, t1, BN);
  conv_kernel<<<256, 128, 0, stream>>>(t1, pwC1B, 64, 1, x0, pwC1S, 64, 2, 64,
                                       y1, BN);
  conv_kernel<<<512, 128, 0, stream>>>(y1, pwC2A, 64, 1, nullptr, nullptr, 0, 0,
                                       128, t2, BN);
  conv_kernel<<<512, 128, 0, stream>>>(t2, pwC2B, 128, 1, y1, pwC2S, 64, 2, 128,
                                       y2, BN);
  conv_kernel<<<1024, 128, 0, stream>>>(y2, pwC3A, 128, 1, nullptr, nullptr, 0,
                                        0, 256, t3, BN);
  conv_kernel<<<1024, 128, 0, stream>>>(t3, pwC3B, 256, 1, y2, pwC3S, 128, 2,
                                        256, y3, BN);

  // --- max-pool + fc ---
  pool_fc_kernel<<<1, 256, 0, stream>>>(y3, fcw, fcb, (float*)d_out);
}
